// GaussianEM_73443940761980
// MI455X (gfx1250) — matrix operand order repaired and checked by reading, not compiled, not run on hardware
//
#include <hip/hip_runtime.h>
#include <hip/hip_bf16.h>

typedef __attribute__((ext_vector_type(16))) _Float16 v16h;
typedef __attribute__((ext_vector_type(8)))  _Float16 v8h;
typedef __attribute__((ext_vector_type(8)))  float    v8f;

#define NSAMP  262144
#define DIM    64
#define NCLS   128
#define LAST   127
#define CHUNK  2048   // samples per workgroup in the segmented-sum kernel

static __device__ __forceinline__ v8f wmma_f16(v16h a, v16h b, v8f c) {
  // v_wmma_f32_16x16x32_f16 : D = A(16x32 f16) * B(32x16 f16) + C(16x16 f32)
  return __builtin_amdgcn_wmma_f32_16x16x32_f16(false, a, false, b, (short)0, c,
                                                false, false);
}

// ---------------------------------------------------------------------------
// Kernel 1: per-class counts + sums via one-hot WMMA GEMM.
//   sums[C,D] += OneHot(target)^T @ X   (X split into f16 hi + f16 residual)
// Block = 256 threads = 8 waves; wave w owns classes [16w, 16w+16).
// Staging: each thread owns 8 contiguous dims of one sample -> 2x b128 global
// loads + 2x b128 LDS stores; next tile prefetched (global_prefetch_b8).
// ---------------------------------------------------------------------------
__global__ __launch_bounds__(256)
void k1_segsum(const float* __restrict__ x, const int* __restrict__ tgt,
               float* __restrict__ sums, float* __restrict__ counts) {
  __shared__ __align__(32) _Float16 sh_hi[32][DIM];
  __shared__ __align__(32) _Float16 sh_lo[32][DIM];
  __shared__ int sh_t[32];
  __shared__ int sh_cnt[NCLS];

  const int tid   = threadIdx.x;
  const int lane  = tid & 31;
  const int wave  = tid >> 5;
  const int cbase = wave << 4;                 // this wave's class-tile base

  const int smp = tid >> 3;                    // sample within 32-tile
  const int dd  = (tid & 7) << 3;              // dim base (0,8,...,56)

  for (int i = tid; i < NCLS; i += 256) sh_cnt[i] = 0;

  v8f acc[4] = {};                             // 4 dim-tiles of 16x16 f32

  const long sbase = (long)blockIdx.x * CHUNK;
  for (int it = 0; it < CHUNK / 32; ++it) {
    __syncthreads();                           // LDS reuse guard
    const long s0 = sbase + (long)it * 32;

    // Stage 32 samples x 64 dims: split f32 -> f16 hi + f16 residual.
    const float4* src = (const float4*)&x[(s0 + smp) * DIM + dd];
    float4 p0 = src[0];
    float4 p1 = src[1];
    if (it + 1 < CHUNK / 32) {                 // prefetch next 32-sample tile
      __builtin_prefetch((const char*)src + (size_t)32 * DIM * sizeof(float), 0, 1);
    }
    v8h hh, hl;
    {
      float e;
      e = p0.x; hh[0] = (_Float16)e; hl[0] = (_Float16)(e - (float)hh[0]);
      e = p0.y; hh[1] = (_Float16)e; hl[1] = (_Float16)(e - (float)hh[1]);
      e = p0.z; hh[2] = (_Float16)e; hl[2] = (_Float16)(e - (float)hh[2]);
      e = p0.w; hh[3] = (_Float16)e; hl[3] = (_Float16)(e - (float)hh[3]);
      e = p1.x; hh[4] = (_Float16)e; hl[4] = (_Float16)(e - (float)hh[4]);
      e = p1.y; hh[5] = (_Float16)e; hl[5] = (_Float16)(e - (float)hh[5]);
      e = p1.z; hh[6] = (_Float16)e; hl[6] = (_Float16)(e - (float)hh[6]);
      e = p1.w; hh[7] = (_Float16)e; hl[7] = (_Float16)(e - (float)hh[7]);
    }
    *(v8h*)&sh_hi[smp][dd] = hh;
    *(v8h*)&sh_lo[smp][dd] = hl;

    if (tid < 32) {
      int t = tgt[s0 + tid];
      sh_t[tid] = t;
      atomicAdd(&sh_cnt[t], 1);
    }
    __syncthreads();

    // Build A = one-hot [16 classes x 32 samples], 16-bit A 16x32 layout:
    //   VGPR v: K = (v>=4?16:0) + (v&3)*2 (+8 for lanes 16..31), M = lane&15.
    const int m    = lane & 15;
    const int koff = (lane >> 4) << 3;
    v16h a;
    #pragma unroll
    for (int v = 0; v < 8; ++v) {
      int kb = ((v >= 4) ? 16 : 0) + ((v & 3) << 1) + koff;
      a[2 * v]     = (sh_t[kb]     == cbase + m) ? (_Float16)1.0f : (_Float16)0.0f;
      a[2 * v + 1] = (sh_t[kb + 1] == cbase + m) ? (_Float16)1.0f : (_Float16)0.0f;
    }

    // B[32x16] per dim-tile: N = lane&15 (dim), K (sample) in same order as A.
    #pragma unroll
    for (int t = 0; t < 4; ++t) {
      v16h bh, bl;
      #pragma unroll
      for (int v = 0; v < 8; ++v) {
        int kb = ((v >= 4) ? 16 : 0) + ((v & 3) << 1) + koff;
        bh[2 * v]     = sh_hi[kb][t * 16 + m];
        bh[2 * v + 1] = sh_hi[kb + 1][t * 16 + m];
        bl[2 * v]     = sh_lo[kb][t * 16 + m];
        bl[2 * v + 1] = sh_lo[kb + 1][t * 16 + m];
      }
      acc[t] = wmma_f16(a, bl, acc[t]);        // residual part
      acc[t] = wmma_f16(a, bh, acc[t]);        // hi part
    }
  }

  // C/D layout: VGPR r -> M = r (+8 for lanes 16..31), N = lane&15.
  const int mrow = (lane >> 4) << 3;
  #pragma unroll
  for (int t = 0; t < 4; ++t) {
    #pragma unroll
    for (int r = 0; r < 8; ++r) {
      int c = cbase + mrow + r;
      int d = t * 16 + (lane & 15);
      atomicAdd(&sums[c * DIM + d], acc[t][r]);
    }
  }

  __syncthreads();
  for (int i = tid; i < NCLS; i += 256) atomicAdd(&counts[i], (float)sh_cnt[i]);
}

// ---------------------------------------------------------------------------
// Kernel 2: means = sums / counts ; s = sums - counts*means (rounding residue)
// ---------------------------------------------------------------------------
__global__ void k2_means(const float* __restrict__ sums,
                         const float* __restrict__ counts,
                         float* __restrict__ means, float* __restrict__ svec) {
  int i = blockIdx.x * blockDim.x + threadIdx.x;   // < NCLS*DIM
  int c = i >> 6;
  float cnt = counts[c];
  float m = sums[i] / cnt;
  means[i] = m;
  svec[i]  = sums[i] - cnt * m;
}

// ---------------------------------------------------------------------------
// Kernel 3: cov_sq_last — only samples of class LAST contribute.
// ---------------------------------------------------------------------------
__global__ void k3_cov(const float* __restrict__ x, const int* __restrict__ tgt,
                       const float* __restrict__ means,
                       const float* __restrict__ svec,
                       const float* __restrict__ counts,
                       float* __restrict__ accum) {
  long i = (long)blockIdx.x * blockDim.x + threadIdx.x;
  if (i >= NSAMP) return;
  if (tgt[i] != LAST) return;
  const float* mrow = &means[LAST * DIM];
  const float* srow = &svec[LAST * DIM];
  const float* xr   = &x[i * DIM];
  float dot = 0.f;
  #pragma unroll
  for (int d = 0; d < DIM; ++d) dot += (xr[d] - mrow[d]) * srow[d];
  float cov = dot / counts[LAST];
  atomicAdd(&accum[0], cov * cov);
}

// ---------------------------------------------------------------------------
// Kernel 4: sum_c ||means[LAST]-mean_param[c]||^2  +  sum(cov_param^2)
// ---------------------------------------------------------------------------
__global__ void k4_terms(const float* __restrict__ means,
                         const float* __restrict__ mean_param,
                         const float* __restrict__ cov_param,
                         float* __restrict__ accum) {
  const int TOT = NCLS * DIM + NCLS * DIM * DIM;   // 8192 + 524288
  float part = 0.f;
  for (int i = blockIdx.x * blockDim.x + threadIdx.x; i < TOT;
       i += gridDim.x * blockDim.x) {
    if (i < NCLS * DIM) {
      int d = i & 63;
      float df = means[LAST * DIM + d] - mean_param[i];
      part += df * df;
    } else {
      float v = cov_param[i - NCLS * DIM];
      part += v * v;
    }
  }
  #pragma unroll
  for (int off = 16; off > 0; off >>= 1) part += __shfl_down(part, off, 32);
  if ((threadIdx.x & 31) == 0) atomicAdd(&accum[1], part);
}

// ---------------------------------------------------------------------------
// Kernel 5: combine into the scalar output.
// ---------------------------------------------------------------------------
__global__ void k5_final(const float* __restrict__ accum, float* __restrict__ out) {
  out[0] = accum[0] + accum[1];
}

extern "C" void kernel_launch(void* const* d_in, const int* in_sizes, int n_in,
                              void* d_out, int out_size, void* d_ws, size_t ws_size,
                              hipStream_t stream) {
  (void)in_sizes; (void)n_in; (void)out_size; (void)ws_size;
  const float* x          = (const float*)d_in[0];
  const int*   tgt        = (const int*)d_in[1];
  const float* mean_param = (const float*)d_in[2];
  const float* cov_param  = (const float*)d_in[3];
  float* out = (float*)d_out;

  float* sums   = (float*)d_ws;             // [NCLS*DIM]
  float* counts = sums + NCLS * DIM;        // [NCLS]
  float* accum  = counts + NCLS;            // [2]  (cov_sq, dist terms)
  float* means  = accum + 2;                // [NCLS*DIM]
  float* svec   = means + NCLS * DIM;       // [NCLS*DIM]

  // Zero the accumulated regions (sums, counts, accum) each call.
  hipMemsetAsync(d_ws, 0, (size_t)(NCLS * DIM + NCLS + 2) * sizeof(float), stream);

  k1_segsum<<<NSAMP / CHUNK, 256, 0, stream>>>(x, tgt, sums, counts);
  k2_means <<<(NCLS * DIM) / 256, 256, 0, stream>>>(sums, counts, means, svec);
  k3_cov   <<<NSAMP / 256, 256, 0, stream>>>(x, tgt, means, svec, counts, accum);
  k4_terms <<<256, 256, 0, stream>>>(means, mean_param, cov_param, accum);
  k5_final <<<1, 1, 0, stream>>>(accum, out);
}
